// SP_loss_14448269984176
// MI455X (gfx1250) — matrix operand order repaired and checked by reading, not compiled, not run on hardware
//
#include <hip/hip_runtime.h>
#include <hip/hip_bf16.h>
#include <math.h>

typedef __attribute__((ext_vector_type(16))) _Float16 v16h;
typedef __attribute__((ext_vector_type(8)))  float    v8f;

#define BB   65536
#define TT   256
#define FF   110          // frequency bins
#define NPAD 224          // 2*110 -> pad to 14 tiles of 16
#define NT   14           // N tiles
#define KC   8            // K chunks of 32
#define ROWS_PER_WG 64    // 4 waves * 16 rows
#define SM_STRIDE 228     // padded LDS row stride (floats)

// ---------------------------------------------------------------------------
// Kernel 1: build fused (hanning * sin/cos) weights, pre-permuted into the
// v_wmma_f32_16x16x32_f16 B-matrix lane layout:
//   element j (0..15) of lane L in chunk (nt,kc) is
//     W[f = nt*16 + (L&15)][k = kc*32 + (L>>4)*8 + (j&7) + (j>>3)*16]
// stored contiguously as 32 bytes per lane -> one aligned b128-pair load.
// ---------------------------------------------------------------------------
__global__ void build_weights(const float* __restrict__ hann,
                              _Float16* __restrict__ wpack) {
    int tid = blockIdx.x * blockDim.x + threadIdx.x;   // 14*8*32 = 3584
    if (tid >= NT * KC * 32) return;
    int lane = tid & 31;
    int kc   = (tid >> 5) & 7;
    int nt   = tid >> 8;
    int f    = nt * 16 + (lane & 15);
    int kbase = kc * 32 + (lane >> 4) * 8;
    _Float16* dst = wpack + (size_t)tid * 16;
#pragma unroll
    for (int j = 0; j < 16; ++j) {
        int k = kbase + (j & 7) + (j >> 3) * 16;
        float w = 0.0f;
        if (f < 2 * FF) {
            int bin = (f < FF) ? f : f - FF;
            float freq = (40.0f + (float)bin) * (1.0f / 60.0f) * (1.0f / 30.0f);
            float a = 2.0f * 3.14159265f * freq * (float)k;
            float v = (f < FF) ? __sinf(a) : __cosf(a);
            w = v * hann[k];
        }
        dst[j] = (_Float16)w;
    }
}

// ---------------------------------------------------------------------------
// Kernel 2: WMMA GEMM (M=65536, N=224, K=256) + fused epilogue.
// Block = 128 threads = 4 waves; each wave owns a 16-row M tile.
// ---------------------------------------------------------------------------
__global__ void __launch_bounds__(128)
gemm_epilogue(const float* __restrict__ wave,
              const float* __restrict__ gt,
              const _Float16* __restrict__ wpack,
              float* __restrict__ out,        // out[0]=loss (written later), out[1..]=idx
              float* __restrict__ partials) {
    __shared__ float sm[ROWS_PER_WG * SM_STRIDE];
    __shared__ float loss_sm[ROWS_PER_WG];

    const int tid     = threadIdx.x;
    const int lane    = tid & 31;
    const int wave_id = tid >> 5;
    const int hi      = lane >> 4;            // lane group 0/1
    const int lo      = lane & 15;

    const int row_base  = blockIdx.x * ROWS_PER_WG + wave_id * 16;
    const int m         = row_base + lo;      // A-matrix row for this lane
    const float* wrow   = wave + (size_t)m * TT;

    v8f acc[NT];
#pragma unroll
    for (int nt = 0; nt < NT; ++nt) acc[nt] = (v8f)(0.0f);

    for (int kc = 0; kc < KC; ++kc) {
        // A fragment: lanes 0-15 cover K = {c..c+7, c+16..c+23}, lanes 16-31
        // cover K = {c+8..c+15, c+24..c+31} within this 32-wide chunk.
        const int k0 = kc * 32 + hi * 8;
        float4 a0 = *(const float4*)(wrow + k0);
        float4 a1 = *(const float4*)(wrow + k0 + 4);
        float4 a2 = *(const float4*)(wrow + k0 + 16);
        float4 a3 = *(const float4*)(wrow + k0 + 20);
        v16h af;
        af[0]  = (_Float16)a0.x; af[1]  = (_Float16)a0.y;
        af[2]  = (_Float16)a0.z; af[3]  = (_Float16)a0.w;
        af[4]  = (_Float16)a1.x; af[5]  = (_Float16)a1.y;
        af[6]  = (_Float16)a1.z; af[7]  = (_Float16)a1.w;
        af[8]  = (_Float16)a2.x; af[9]  = (_Float16)a2.y;
        af[10] = (_Float16)a2.z; af[11] = (_Float16)a2.w;
        af[12] = (_Float16)a3.x; af[13] = (_Float16)a3.y;
        af[14] = (_Float16)a3.z; af[15] = (_Float16)a3.w;

#pragma unroll
        for (int nt = 0; nt < NT; ++nt) {
            // B fragment: 32 bytes/lane, pre-permuted, resident in L2
            v16h bf = *(const v16h*)(wpack + (((size_t)nt * KC + kc) * 32 + lane) * 16);
            acc[nt] = __builtin_amdgcn_wmma_f32_16x16x32_f16(
                false, af, false, bf, (short)0, acc[nt], false, false);
        }
    }

    // C/D layout: VGPR r, lanes 0-15 -> M=r, lanes 16-31 -> M=8+r; N = lane&15.
#pragma unroll
    for (int nt = 0; nt < NT; ++nt) {
#pragma unroll
        for (int r = 0; r < 8; ++r) {
            sm[(wave_id * 16 + hi * 8 + r) * SM_STRIDE + nt * 16 + lo] = acc[nt][r];
        }
    }
    __syncthreads();

    // Epilogue: one thread per output row.
    if (tid < ROWS_PER_WG) {
        float* row = sm + tid * SM_STRIDE;
        float bestp = -INFINITY;
        int   besti = 0;
#pragma unroll 2
        for (int f = 0; f < FF; ++f) {
            float s = row[f];
            float c = row[FF + f];
            float p = fmaf(s, s, c * c);
            row[f] = p;                       // overwrite s slot with power
            if (p > bestp) { bestp = p; besti = f; }
        }
        float sum = 0.0f;
#pragma unroll 2
        for (int f = 0; f < FF; ++f) sum += expf(row[f] - bestp);

        int grow = blockIdx.x * ROWS_PER_WG + tid;
        float g = gt[grow];
        g = fminf(fmaxf(g, 40.0f), 149.0f);
        int target = (int)g - 40;

        float lse = bestp + logf(sum);
        loss_sm[tid] = lse - row[target];     // -logp[target]
        out[1 + grow] = (float)(besti + 40);  // argmax + LOW
    }
    __syncthreads();

    if (tid == 0) {
        float t = 0.0f;
        for (int i = 0; i < ROWS_PER_WG; ++i) t += loss_sm[i];   // deterministic
        partials[blockIdx.x] = t;
    }
}

// ---------------------------------------------------------------------------
// Kernel 3: deterministic reduction of per-WG partials -> mean loss.
// ---------------------------------------------------------------------------
__global__ void finalize(const float* __restrict__ partials,
                         float* __restrict__ out) {
    __shared__ float sm[256];
    int t = threadIdx.x;
    float s = partials[t] + partials[t + 256] + partials[t + 512] + partials[t + 768];
    sm[t] = s;
    __syncthreads();
    for (int off = 128; off > 0; off >>= 1) {
        if (t < off) sm[t] += sm[t + off];
        __syncthreads();
    }
    if (t == 0) out[0] = sm[0] * (1.0f / (float)BB);
}

extern "C" void kernel_launch(void* const* d_in, const int* in_sizes, int n_in,
                              void* d_out, int out_size, void* d_ws, size_t ws_size,
                              hipStream_t stream) {
    const float* wave = (const float*)d_in[0];   // [B,T] f32
    const float* gt   = (const float*)d_in[1];   // [B]   f32
    const float* hann = (const float*)d_in[2];   // [T]   f32
    float* out = (float*)d_out;                  // [1 + B]

    // d_ws layout: [0, 114688) packed f16 weights; then 1024 f32 partials.
    _Float16* wpack   = (_Float16*)d_ws;
    float*    partials = (float*)((char*)d_ws + (size_t)NT * KC * 32 * 16 * sizeof(_Float16));

    build_weights<<<NT, 256, 0, stream>>>(hann, wpack);

    dim3 grid(BB / ROWS_PER_WG);                 // 1024 workgroups
    dim3 block(128);                             // 4 waves (wave32)
    gemm_epilogue<<<grid, block, 0, stream>>>(wave, gt, wpack, out, partials);

    finalize<<<1, 256, 0, stream>>>(partials, out);
}
